// GraphIpaFrameDenoisingLayer_20117626814820
// MI455X (gfx1250) — matrix operand order repaired and checked by reading, not compiled, not run on hardware
//
#include <hip/hip_runtime.h>

// ---------------------------------------------------------------------------
// GraphIpaFrameDenoisingLayer for MI455X (gfx1250, wave32, WMMA)
// GEMMs: v_wmma_f32_16x16x32_bf16, 32x32 tile per wave (4 WMMA/K-step),
// weights pre-shuffled to tiled col-major bf16, W tiles staged to LDS via
// async global->LDS DMA when the toolchain exposes the gfx1250 builtin.
// ---------------------------------------------------------------------------

typedef __attribute__((ext_vector_type(16))) __bf16          v16bf;
typedef __attribute__((ext_vector_type(16))) unsigned short  v16us;
typedef __attribute__((ext_vector_type(8)))  unsigned short  v8us;
typedef __attribute__((ext_vector_type(4)))  unsigned short  v4us;
typedef __attribute__((ext_vector_type(8)))  float           v8f;

#if defined(__has_builtin)
#if __has_builtin(__builtin_amdgcn_global_load_async_to_lds_b128)
#define HAVE_ASYNC_LDS 1
#endif
#if __has_builtin(__builtin_amdgcn_s_wait_asynccnt)
#define HAVE_WAIT_ASYNC 1
#endif
#endif

#if defined(HAVE_ASYNC_LDS)
// exact parameter types per hipcc diagnostic: pointer to <4 x i32> in AS(1)/AS(3)
typedef int b128i __attribute__((vector_size(16)));
typedef b128i __attribute__((address_space(1)))* as1_b128;
typedef b128i __attribute__((address_space(3)))* as3_b128;
#endif

#define CS   256
#define CZ   64
#define CH   16
#define NH   8
#define PQN  4
#define PVN  8
#define HIDW 320
#define CATW 896   // H*(C_H + PV*3 + PV + C_Z) = 128+192+64+512

__device__ __forceinline__ unsigned short f2bf(float f) {
  unsigned int u = __float_as_uint(f);
  u += 0x7FFFu + ((u >> 16) & 1u);          // round-to-nearest-even
  return (unsigned short)(u >> 16);
}

__device__ __forceinline__ void wait_async_lds() {
#if defined(HAVE_ASYNC_LDS)
#if defined(HAVE_WAIT_ASYNC)
  __builtin_amdgcn_s_wait_asynccnt(0);
#else
  asm volatile("s_wait_asynccnt 0x0" ::: "memory");
#endif
#endif
}

// ---------------------------------------------------------------------------
// Weight shuffle: f32 row-major [K,Nc] -> bf16 tiled col-major:
//   tile (nt = c/32, kk = k/32); element -> out[((nt*(K/32)+kk)*32 + c%32)*32 + k%32]
// Each 32x32 tile is a contiguous 2KB block (async-DMA friendly), and a
// B-fragment is two 8-half runs per lane (k 8g.. and 16+8g..) (ds_load_b128 x2).
// ---------------------------------------------------------------------------
__global__ void k_wtile(const float* __restrict__ w, unsigned short* __restrict__ o,
                        int K, int Nc) {
  int i = blockIdx.x * 256 + threadIdx.x;
  if (i >= K * Nc) return;
  int k = i / Nc, c = i - k * Nc;
  int nt = c >> 5, cc = c & 31, kk = k >> 5, kr = k & 31;
  o[(((size_t)nt * (K >> 5) + kk) * 32 + cc) * 32 + kr] = f2bf(w[i]);
}

// ---------------------------------------------------------------------------
// GEMM: Y[M,Nc] = act( (A (+A2)) @ W + bias ),  Wt = tiled bf16 (see k_wtile)
// grid = (M/32, Nc/32), block = 32 (one wave). 4 WMMAs per K-step.
// ---------------------------------------------------------------------------
__global__ void k_gemm(const float* __restrict__ A, const float* __restrict__ A2,
                       const unsigned short* __restrict__ Wt,
                       const float* __restrict__ bias,
                       float* __restrict__ Y, int M, int K, int Nc, int relu)
{
  __shared__ __align__(16) unsigned short sA[32 * 32];  // A tile, row-major
  __shared__ __align__(16) unsigned short sB[32 * 32];  // W tile, col-major
  const int lane = threadIdx.x;
  const int g = lane >> 4, ml = lane & 15;
  const int row0 = blockIdx.x * 32, col0 = blockIdx.y * 32;
  const int ktiles = K >> 5;
  const unsigned short* wpanel = Wt + (size_t)blockIdx.y * ktiles * 1024;
  v8f acc00 = {}, acc01 = {}, acc10 = {}, acc11 = {};

  for (int kk = 0; kk < ktiles; ++kk) {
    const unsigned short* wtile = wpanel + (size_t)kk * 1024;
    // ---- stage W tile (2KB contiguous) ----
#if defined(HAVE_ASYNC_LDS)
#pragma unroll
    for (int it = 0; it < 4; ++it) {
      as1_b128 gsrc = (as1_b128)(unsigned long long)(wtile + it * 256 + lane * 8);
      as3_b128 ldst = (as3_b128)(unsigned int)(unsigned long long)(&sB[it * 256 + lane * 8]);
      __builtin_amdgcn_global_load_async_to_lds_b128(gsrc, ldst, 0, 0);
    }
#else
    if (kk + 1 < ktiles)
      __builtin_prefetch(wtile + 1024 + lane * 8, 0, 1);
#pragma unroll
    for (int it = 0; it < 4; ++it) {
      int i = it * 256 + lane * 8;
      *(v8us*)&sB[i] = *(const v8us*)&wtile[i];
    }
#endif
    // ---- stage A tile 32x32 (f32 -> bf16) via float4 loads ----
    int k0 = kk * 32;
#pragma unroll
    for (int it = 0; it < 8; ++it) {
      int idx = it * 32 + lane;            // 256 float4 chunks
      int r = idx >> 3, c4 = (idx & 7) * 4;
      size_t off = (size_t)(row0 + r) * K + k0 + c4;
      float4 v = *(const float4*)(A + off);
      if (A2) {
        float4 w2 = *(const float4*)(A2 + off);
        v.x += w2.x; v.y += w2.y; v.z += w2.z; v.w += w2.w;
      }
      v4us p = { f2bf(v.x), f2bf(v.y), f2bf(v.z), f2bf(v.w) };
      *(v4us*)&sA[r * 32 + c4] = p;
    }
    __syncthreads();
    wait_async_lds();

    // ---- fragments (each = 2x ds_load_b128) ----
    v8us a0l = *(const v8us*)&sA[ml * 32 + g * 8];
    v8us a0h = *(const v8us*)&sA[ml * 32 + 16 + g * 8];
    v8us a1l = *(const v8us*)&sA[(16 + ml) * 32 + g * 8];
    v8us a1h = *(const v8us*)&sA[(16 + ml) * 32 + 16 + g * 8];
    v8us b0l = *(const v8us*)&sB[ml * 32 + g * 8];
    v8us b0h = *(const v8us*)&sB[ml * 32 + 16 + g * 8];
    v8us b1l = *(const v8us*)&sB[(16 + ml) * 32 + g * 8];
    v8us b1h = *(const v8us*)&sB[(16 + ml) * 32 + 16 + g * 8];
    v16us ua0 = __builtin_shufflevector(a0l, a0h, 0,1,2,3,4,5,6,7,8,9,10,11,12,13,14,15);
    v16us ua1 = __builtin_shufflevector(a1l, a1h, 0,1,2,3,4,5,6,7,8,9,10,11,12,13,14,15);
    v16us ub0 = __builtin_shufflevector(b0l, b0h, 0,1,2,3,4,5,6,7,8,9,10,11,12,13,14,15);
    v16us ub1 = __builtin_shufflevector(b1l, b1h, 0,1,2,3,4,5,6,7,8,9,10,11,12,13,14,15);
    v16bf fa0 = __builtin_bit_cast(v16bf, ua0);
    v16bf fa1 = __builtin_bit_cast(v16bf, ua1);
    v16bf fb0 = __builtin_bit_cast(v16bf, ub0);
    v16bf fb1 = __builtin_bit_cast(v16bf, ub1);
    acc00 = __builtin_amdgcn_wmma_f32_16x16x32_bf16(false, fa0, false, fb0, (short)0, acc00, false, false);
    acc01 = __builtin_amdgcn_wmma_f32_16x16x32_bf16(false, fa0, false, fb1, (short)0, acc01, false, false);
    acc10 = __builtin_amdgcn_wmma_f32_16x16x32_bf16(false, fa1, false, fb0, (short)0, acc10, false, false);
    acc11 = __builtin_amdgcn_wmma_f32_16x16x32_bf16(false, fa1, false, fb1, (short)0, acc11, false, false);
    __syncthreads();
  }

  int c0 = col0 + ml, c1 = col0 + 16 + ml;
  float bi0 = bias[c0], bi1 = bias[c1];
#pragma unroll
  for (int r = 0; r < 8; ++r) {
    int rowA = row0 + r + 8 * g;
    int rowB = row0 + 16 + r + 8 * g;
    float v00 = acc00[r] + bi0, v01 = acc01[r] + bi1;
    float v10 = acc10[r] + bi0, v11 = acc11[r] + bi1;
    if (relu) {
      v00 = fmaxf(v00, 0.f); v01 = fmaxf(v01, 0.f);
      v10 = fmaxf(v10, 0.f); v11 = fmaxf(v11, 0.f);
    }
    Y[(size_t)rowA * Nc + c0] = v00;
    Y[(size_t)rowA * Nc + c1] = v01;
    Y[(size_t)rowB * Nc + c0] = v10;
    Y[(size_t)rowB * Nc + c1] = v11;
  }
}

// ---------------------------------------------------------------------------
__global__ void k_fill(float* p, float v, int nelem) {
  int i = blockIdx.x * 256 + threadIdx.x;
  if (i < nelem) p[i] = v;
}

__global__ void k_masks(const int* __restrict__ xm, const int* __restrict__ nm,
                        float* __restrict__ valid, float* __restrict__ nmf, int n) {
  int i = blockIdx.x * 256 + threadIdx.x;
  if (i < n) { valid[i] = 1.f - (float)xm[i]; nmf[i] = (float)nm[i]; }
}

__global__ void k_rowscale(const float* __restrict__ x, const float* __restrict__ sc,
                           float* __restrict__ y, int D, int n) {
  int i = blockIdx.x * 256 + threadIdx.x;
  if (i < n * D) y[i] = x[i] * sc[i / D];
}

// local -> global frame: p = R*p + t (in place; one thread per point)
__global__ void k_frame_fwd(const float* __restrict__ rots, const float* __restrict__ trans,
                            float* __restrict__ pts, int P, int n) {
  int idx = blockIdx.x * 256 + threadIdx.x;
  if (idx >= n * P) return;
  int nd = idx / P;
  float* p = pts + (size_t)idx * 3;
  float x = p[0], y = p[1], z = p[2];
  const float* R = rots + nd * 9;
  p[0] = R[0]*x + R[1]*y + R[2]*z + trans[nd*3+0];
  p[1] = R[3]*x + R[4]*y + R[5]*z + trans[nd*3+1];
  p[2] = R[6]*x + R[7]*y + R[8]*z + trans[nd*3+2];
}

// pair bias b = z @ Wb + bb   (E x 8), scalar (Nc too small for a WMMA tile)
__global__ void k_edge_bias(const float* __restrict__ z, const float* __restrict__ w,
                            const float* __restrict__ b, float* __restrict__ out, int E) {
  int i = blockIdx.x * 256 + threadIdx.x;
  if (i >= E * NH) return;
  int e = i >> 3, hh = i & 7;
  float acc = b[hh];
  const float* zr = z + (size_t)e * CZ;
#pragma unroll 8
  for (int c = 0; c < CZ; ++c) acc += zr[c] * w[c * NH + hh];
  out[i] = acc;
}

__global__ void k_logits(const float* __restrict__ q, const float* __restrict__ k,
                         const float* __restrict__ qp, const float* __restrict__ kp,
                         const float* __restrict__ bedge, const float* __restrict__ headw,
                         const int* __restrict__ src, const int* __restrict__ dst,
                         const float* __restrict__ valid, float* __restrict__ a, int E)
{
  int i = blockIdx.x * 256 + threadIdx.x;
  if (i >= E * NH) return;
  int e = i >> 3, h = i & 7;
  int s_ = src[e], d_ = dst[e];
  const float* qr = q + ((size_t)d_ * NH + h) * CH;
  const float* kr = k + ((size_t)s_ * NH + h) * CH;
  float dot = 0.f;
#pragma unroll
  for (int c = 0; c < CH; ++c) dot += qr[c] * kr[c];
  float aa = dot * 0.14433756729740643f            // sqrt(1/48)
           + 0.5773502691896258f * bedge[i];       // sqrt(1/3)
  float hw = log1pf(expf(headw[h])) * 0.13608276348795434f;  // sqrt(1/54)
  const float* qpr = qp + ((size_t)d_ * NH + h) * PQN * 3;
  const float* kpr = kp + ((size_t)s_ * NH + h) * PQN * 3;
  float d2 = 0.f;
#pragma unroll
  for (int p = 0; p < PQN; ++p) {
    float dx = qpr[p*3+0] - kpr[p*3+0];
    float dy = qpr[p*3+1] - kpr[p*3+1];
    float dz = qpr[p*3+2] - kpr[p*3+2];
    d2 += dx*dx + dy*dy + dz*dz;
  }
  aa -= 0.5f * hw * d2;
  aa += -1e9f * (1.f - valid[s_] * valid[d_]);
  a[i] = aa;
}

// monotonic float<->uint key for atomicMax segment-max
__device__ __forceinline__ unsigned int fkey(float f) {
  unsigned int u = __float_as_uint(f);
  return (u & 0x80000000u) ? ~u : (u | 0x80000000u);
}
__device__ __forceinline__ float finv(unsigned int kk) {
  unsigned int u = (kk & 0x80000000u) ? (kk & 0x7FFFFFFFu) : ~kk;
  return __uint_as_float(u);
}

__global__ void k_segmax(const float* __restrict__ a, const int* __restrict__ dst,
                         unsigned int* __restrict__ amax, int E) {
  int i = blockIdx.x * 256 + threadIdx.x;
  if (i >= E * NH) return;
  int e = i >> 3, h = i & 7;
  atomicMax(&amax[(size_t)dst[e] * NH + h], fkey(a[i]));
}

__global__ void k_expsum(float* __restrict__ a, const int* __restrict__ dst,
                         const unsigned int* __restrict__ amax,
                         float* __restrict__ asum, int E) {
  int i = blockIdx.x * 256 + threadIdx.x;
  if (i >= E * NH) return;
  int e = i >> 3, h = i & 7;
  int d_ = dst[e];
  float ex = expf(a[i] - finv(amax[(size_t)d_ * NH + h]));
  a[i] = ex;
  atomicAdd(&asum[(size_t)d_ * NH + h], ex);
}

// one block (256 thr) per edge: scatter attn*v / attn*v_pts / attn*z into cat rows
__global__ void k_scatter(const float* __restrict__ ex, const float* __restrict__ asum,
                          const int* __restrict__ src, const int* __restrict__ dst,
                          const float* __restrict__ v, const float* __restrict__ vpg,
                          const float* __restrict__ z, float* __restrict__ cat, int E)
{
  int e = blockIdx.x;
  if (e >= E) return;
  __shared__ float at[NH];
  int t = threadIdx.x;
  int s_ = src[e], d_ = dst[e];
  if (t < NH) at[t] = ex[(size_t)e * NH + t] / (asum[(size_t)d_ * NH + t] + 1e-9f);
  __syncthreads();
  float* crow = cat + (size_t)d_ * CATW;
  for (int j = t; j < 832; j += 256) {
    if (j < 128) {                                  // o: [H*C_H]
      atomicAdd(&crow[j], at[j >> 4] * v[(size_t)s_ * 128 + j]);
    } else if (j < 320) {                           // o_pt (global): [H*PV*3]
      int jj = j - 128;
      atomicAdd(&crow[128 + jj], at[jj / 24] * vpg[(size_t)s_ * 192 + jj]);
    } else {                                        // o_pair: [H*C_Z] at +384
      int jj = j - 320;
      atomicAdd(&crow[384 + jj], at[jj >> 6] * z[(size_t)e * CZ + (jj & 63)]);
    }
  }
}

// transform accumulated o_pt back to local frame + write norms; 1 thr per (n,h,p)
__global__ void k_finalize_pts(const float* __restrict__ rots, const float* __restrict__ trans,
                               float* __restrict__ cat, int n) {
  int idx = blockIdx.x * 256 + threadIdx.x;
  if (idx >= n * 64) return;
  int nd = idx >> 6, hp = idx & 63;
  float* c = cat + (size_t)nd * CATW + 128 + hp * 3;
  float x = c[0] - trans[nd*3+0];
  float y = c[1] - trans[nd*3+1];
  float z = c[2] - trans[nd*3+2];
  const float* R = rots + nd * 9;
  float lx = R[0]*x + R[3]*y + R[6]*z;  // R^T
  float ly = R[1]*x + R[4]*y + R[7]*z;
  float lz = R[2]*x + R[5]*y + R[8]*z;
  c[0] = lx; c[1] = ly; c[2] = lz;
  cat[(size_t)nd * CATW + 320 + hp] = sqrtf(lx*lx + ly*ly + lz*lz + 1e-8f);
}

// out = LN(base + upd*scale?) ; one row per block, blockDim == D (64 or 256)
__global__ void k_res_ln(const float* __restrict__ base, const float* __restrict__ upd,
                         const float* __restrict__ scale, const float* __restrict__ g,
                         const float* __restrict__ b, float* __restrict__ out, int D)
{
  __shared__ float red[256];
  __shared__ float mu_s, rs_s;
  int row = blockIdx.x, t = threadIdx.x;
  float x = base[(size_t)row * D + t];
  if (upd) {
    float u = upd[(size_t)row * D + t];
    if (scale) u *= scale[row];
    x += u;
  }
  red[t] = x; __syncthreads();
  for (int s = D >> 1; s > 0; s >>= 1) { if (t < s) red[t] += red[t + s]; __syncthreads(); }
  if (t == 0) mu_s = red[0] / D; __syncthreads();
  float d = x - mu_s;
  red[t] = d * d; __syncthreads();
  for (int s = D >> 1; s > 0; s >>= 1) { if (t < s) red[t] += red[t + s]; __syncthreads(); }
  if (t == 0) rs_s = rsqrtf(red[0] / D + 1e-5f); __syncthreads();
  out[(size_t)row * D + t] = d * rs_s * g[t] + b[t];
}

// h = [z | nsmall[src] | nsmall[dst]]   (E x 320)
__global__ void k_build_h(const float* __restrict__ z, const float* __restrict__ ns,
                          const int* __restrict__ src, const int* __restrict__ dst,
                          float* __restrict__ h, int E) {
  int idx = blockIdx.x * 256 + threadIdx.x;
  if (idx >= E * HIDW) return;
  int e = idx / HIDW, c = idx - e * HIDW;
  float v;
  if (c < CZ)        v = z[(size_t)e * CZ + c];
  else if (c < 192)  v = ns[(size_t)src[e] * 128 + (c - 64)];
  else               v = ns[(size_t)dst[e] * 128 + (c - 192)];
  h[idx] = v;
}

// backbone update: one thread per node (6-dim linear + quat compose)
__global__ void k_backbone(const float* __restrict__ s, const float* __restrict__ nmf,
                           const float* __restrict__ w, const float* __restrict__ bias,
                           const float* __restrict__ rots, const float* __restrict__ trans,
                           float* __restrict__ orot, float* __restrict__ otr, int n)
{
  int nd = blockIdx.x * 64 + threadIdx.x;
  if (nd >= n) return;
  float m = nmf[nd];
  const float* sr = s + (size_t)nd * CS;
  float u[6];
#pragma unroll
  for (int j = 0; j < 6; ++j) {
    float acc = bias[j];
    for (int c = 0; c < CS; ++c) acc += (sr[c] * m) * w[c * 6 + j];
    u[j] = acc * m;
  }
  float qw = 1.f, qx = u[0], qy = u[1], qz = u[2];
  float inv = rsqrtf(qw*qw + qx*qx + qy*qy + qz*qz);
  qw *= inv; qx *= inv; qy *= inv; qz *= inv;
  float Q[9] = {1-2*(qy*qy+qz*qz), 2*(qx*qy-qw*qz), 2*(qx*qz+qw*qy),
                2*(qx*qy+qw*qz), 1-2*(qx*qx+qz*qz), 2*(qy*qz-qw*qx),
                2*(qx*qz-qw*qy), 2*(qy*qz+qw*qx), 1-2*(qx*qx+qy*qy)};
  const float* R = rots + nd * 9;
  float* O = orot + nd * 9;
#pragma unroll
  for (int i = 0; i < 3; ++i)
#pragma unroll
    for (int j = 0; j < 3; ++j)
      O[i*3+j] = R[i*3+0]*Q[0*3+j] + R[i*3+1]*Q[1*3+j] + R[i*3+2]*Q[2*3+j];
#pragma unroll
  for (int i = 0; i < 3; ++i)
    otr[nd*3+i] = trans[nd*3+i] + R[i*3+0]*u[3] + R[i*3+1]*u[4] + R[i*3+2]*u[5];
}

// ---------------------------------------------------------------------------
// Host orchestration
// ---------------------------------------------------------------------------
// Param flattening (jax pytree = sorted dict keys) starting at d_in[9]:
//   attn_seq@9, attn_spatial@26 (each 17 leaves:
//     b.b,b.w,head_w,k.b,k.w,k_pts.b,k_pts.w,out.b,out.w,q.b,q.w,
//     q_pts.b,q_pts.w,v.b,v.w,v_pts.b,v_pts.w)
//   bb.b@43, bb.w@44
//   edge_tr@45 (final.b,final.w,init.b,init.w,ln.b,ln.g,t1.b,t1.w,t2.b,t2.w)
//   ln_s1.b@55,.g@56  ln_s2.b@57,.g@58
//   nt1.b@59,.w@60  nt2.b@61,.w@62  nt3.b@63,.w@64  nt_ln.b@65,.g@66
//   seq_edge_tr@67

extern "C" void kernel_launch(void* const* d_in, const int* in_sizes, int n_in,
                              void* d_out_v, int out_size, void* d_ws, size_t ws_size,
                              hipStream_t stream)
{
  const float* node_f  = (const float*)d_in[0];
  const float* rots    = (const float*)d_in[1];
  const float* trans   = (const float*)d_in[2];
  const float* z_sp    = (const float*)d_in[3];
  const int*   e_sp    = (const int*)d_in[4];
  const float* z_seq   = (const float*)d_in[5];
  const int*   e_seq   = (const int*)d_in[6];
  const int*   x_mask  = (const int*)d_in[7];
  const int*   n_mask  = (const int*)d_in[8];
  const int n    = in_sizes[0] / CS;
  const int Esp  = in_sizes[4] / 2;
  const int Eseq = in_sizes[6] / 2;
  const int Emax = Esp > Eseq ? Esp : Eseq;

  const int P_ASEQ = 9, P_ASP = 26, P_BB_B = 43, P_BB_W = 44, P_ETR = 45,
            P_LN1B = 55, P_LN1G = 56, P_LN2B = 57, P_LN2G = 58,
            P_NT1B = 59, P_NT1W = 60, P_NT2B = 61, P_NT2W = 62,
            P_NT3B = 63, P_NT3W = 64, P_NTLB = 65, P_NTLG = 66, P_SETR = 67;
  auto P = [&](int i) { return (const float*)d_in[i]; };

  // bump allocator over workspace
  char* wsb = (char*)d_ws;
  size_t off = 0;
  auto alloc = [&](size_t bytes) -> void* {
    void* p = wsb + off;
    off = (off + bytes + 255) & ~(size_t)255;
    return p;
  };
  // convert+shuffle one weight to tiled bf16
  auto cvt = [&](int pi, int K, int Nc) -> unsigned short* {
    int nelem = K * Nc;
    unsigned short* dst = (unsigned short*)alloc((size_t)nelem * 2);
    k_wtile<<<(nelem + 255) / 256, 256, 0, stream>>>(P(pi), dst, K, Nc);
    return dst;
  };

  // --- bf16 tiled weights ---
  struct IpaW { unsigned short *q, *k, *v, *qp, *kp, *vp, *out; };
  auto cvt_ipa = [&](int base) {
    IpaW w;
    w.q   = cvt(base + 10, CS, 128);
    w.k   = cvt(base + 4,  CS, 128);
    w.v   = cvt(base + 14, CS, 128);
    w.qp  = cvt(base + 12, CS, 96);
    w.kp  = cvt(base + 6,  CS, 96);
    w.vp  = cvt(base + 16, CS, 192);
    w.out = cvt(base + 8,  CATW, CS);
    return w;
  };
  struct EtrW { unsigned short *init, *t1, *t2, *fin; };
  auto cvt_etr = [&](int base) {
    EtrW w;
    w.fin  = cvt(base + 1, HIDW, CZ);
    w.init = cvt(base + 3, CS, 128);
    w.t1   = cvt(base + 7, HIDW, HIDW);
    w.t2   = cvt(base + 9, HIDW, HIDW);
    return w;
  };
  IpaW wip_sp  = cvt_ipa(P_ASP);
  IpaW wip_seq = cvt_ipa(P_ASEQ);
  EtrW wet_sp  = cvt_etr(P_ETR);
  EtrW wet_seq = cvt_etr(P_SETR);
  unsigned short* w_nt1 = cvt(P_NT1W, CS, CS);
  unsigned short* w_nt2 = cvt(P_NT2W, CS, CS);
  unsigned short* w_nt3 = cvt(P_NT3W, CS, CS);

  // --- activations / scratch ---
  float* valid = (float*)alloc((size_t)n * 4);
  float* nmf   = (float*)alloc((size_t)n * 4);
  float* qb    = (float*)alloc((size_t)n * 128 * 4);
  float* kb    = (float*)alloc((size_t)n * 128 * 4);
  float* vb    = (float*)alloc((size_t)n * 128 * 4);
  float* qp    = (float*)alloc((size_t)n * 96 * 4);
  float* kp    = (float*)alloc((size_t)n * 96 * 4);
  float* vp    = (float*)alloc((size_t)n * 192 * 4);
  float* bedge = (float*)alloc((size_t)Emax * NH * 4);
  float* alog  = (float*)alloc((size_t)Emax * NH * 4);
  unsigned int* amax = (unsigned int*)alloc((size_t)n * NH * 4);
  float* asum  = (float*)alloc((size_t)n * NH * 4);
  float* cat   = (float*)alloc((size_t)n * CATW * 4);
  float* upd   = (float*)alloc((size_t)n * CS * 4);
  float* s1    = (float*)alloc((size_t)n * CS * 4);
  float* s2    = (float*)alloc((size_t)n * CS * 4);
  float* tbuf  = (float*)alloc((size_t)n * CS * 4);
  float* nsm   = (float*)alloc((size_t)n * 128 * 4);
  float* hbuf  = (float*)alloc((size_t)Emax * HIDW * 4);
  float* t1buf = (float*)alloc((size_t)Emax * HIDW * 4);
  float* t2buf = (float*)alloc((size_t)Emax * HIDW * 4);

  float* out_s   = (float*)d_out_v;
  float* out_rot = out_s   + (size_t)n * CS;
  float* out_tr  = out_rot + (size_t)n * 9;
  float* out_z   = out_tr  + (size_t)n * 3;
  float* out_zs  = out_z   + (size_t)Esp * CZ;
  float* out_kl  = out_zs  + (size_t)Eseq * CZ;

  auto gemm = [&](const float* A, const float* A2, const unsigned short* W,
                  const float* bias, float* Y, int M, int K, int Nc, int relu) {
    dim3 g(M / 32, Nc / 32);
    k_gemm<<<g, 32, 0, stream>>>(A, A2, W, bias, Y, M, K, Nc, relu);
  };
  auto fill = [&](float* p, float v, size_t ne) {
    k_fill<<<(int)((ne + 255) / 256), 256, 0, stream>>>(p, v, (int)ne);
  };

  k_masks<<<(n + 255) / 256, 256, 0, stream>>>(x_mask, n_mask, valid, nmf, n);

  auto run_ipa = [&](const float* s_in, const float* z, const int* eidx, int E,
                     int base, const IpaW& w, float* upd_out) {
    const int* src = eidx;
    const int* dst = eidx + E;
    gemm(s_in, nullptr, w.q,  P(base + 9),  qb, n, CS, 128, 0);
    gemm(s_in, nullptr, w.k,  P(base + 3),  kb, n, CS, 128, 0);
    gemm(s_in, nullptr, w.v,  P(base + 13), vb, n, CS, 128, 0);
    gemm(s_in, nullptr, w.qp, P(base + 11), qp, n, CS, 96, 0);
    gemm(s_in, nullptr, w.kp, P(base + 5),  kp, n, CS, 96, 0);
    gemm(s_in, nullptr, w.vp, P(base + 15), vp, n, CS, 192, 0);
    k_frame_fwd<<<(n * 32 + 255) / 256, 256, 0, stream>>>(rots, trans, qp, 32, n);
    k_frame_fwd<<<(n * 32 + 255) / 256, 256, 0, stream>>>(rots, trans, kp, 32, n);
    k_frame_fwd<<<(n * 64 + 255) / 256, 256, 0, stream>>>(rots, trans, vp, 64, n);
    k_edge_bias<<<(E * NH + 255) / 256, 256, 0, stream>>>(z, P(base + 1), P(base + 0), bedge, E);
    k_logits<<<(E * NH + 255) / 256, 256, 0, stream>>>(qb, kb, qp, kp, bedge,
                                                       P(base + 2), src, dst, valid, alog, E);
    fill((float*)amax, 0.f, (size_t)n * NH);   // == fkey floor
    fill(asum, 0.f, (size_t)n * NH);
    fill(cat, 0.f, (size_t)n * CATW);
    k_segmax<<<(E * NH + 255) / 256, 256, 0, stream>>>(alog, dst, amax, E);
    k_expsum<<<(E * NH + 255) / 256, 256, 0, stream>>>(alog, dst, amax, asum, E);
    k_scatter<<<E, 256, 0, stream>>>(alog, asum, src, dst, vb, vp, z, cat, E);
    k_finalize_pts<<<(n * 64 + 255) / 256, 256, 0, stream>>>(rots, trans, cat, n);
    gemm(cat, nullptr, w.out, P(base + 7), upd_out, n, CATW, CS, 0);
  };

  // --- IPA spatial + LN ---
  run_ipa(node_f, z_sp, e_sp, Esp, P_ASP, wip_sp, upd);
  k_res_ln<<<n, CS, 0, stream>>>(node_f, upd, valid, P(P_LN1G), P(P_LN1B), s1, CS);
  // --- IPA seq + LN ---
  run_ipa(s1, z_seq, e_seq, Eseq, P_ASEQ, wip_seq, upd);
  k_res_ln<<<n, CS, 0, stream>>>(s1, upd, valid, P(P_LN2G), P(P_LN2B), s2, CS);

  // --- node transition ---
  gemm(s2,   nullptr, w_nt1, P(P_NT1B), tbuf, n, CS, CS, 1);
  gemm(tbuf, nullptr, w_nt2, P(P_NT2B), s1,   n, CS, CS, 1);   // reuse s1
  gemm(s1,   nullptr, w_nt3, P(P_NT3B), upd,  n, CS, CS, 0);
  k_res_ln<<<n, CS, 0, stream>>>(s2, upd, nullptr, P(P_NTLG), P(P_NTLB), tbuf, CS);
  k_rowscale<<<(n * CS + 255) / 256, 256, 0, stream>>>(tbuf, valid, out_s, CS, n);

  // --- backbone update ---
  k_backbone<<<(n + 63) / 64, 64, 0, stream>>>(out_s, nmf, P(P_BB_W), P(P_BB_B),
                                               rots, trans, out_rot, out_tr, n);

  // --- edge transitions ---
  auto run_etr = [&](const float* z, const int* eidx, int E, int base,
                     const EtrW& w, float* z_out) {
    const int* src = eidx;
    const int* dst = eidx + E;
    gemm(out_s, nullptr, w.init, P(base + 2), nsm, n, CS, 128, 0);
    k_build_h<<<(E * HIDW + 255) / 256, 256, 0, stream>>>(z, nsm, src, dst, hbuf, E);
    gemm(hbuf,  nullptr, w.t1, P(base + 6), t1buf, E, HIDW, HIDW, 1);
    gemm(t1buf, nullptr, w.t2, P(base + 8), t2buf, E, HIDW, HIDW, 1);
    gemm(t2buf, hbuf,    w.fin, P(base + 0), t1buf, E, HIDW, CZ, 0);  // (t+h)@Wf
    k_res_ln<<<E, CZ, 0, stream>>>(t1buf, nullptr, nullptr, P(base + 5), P(base + 4), z_out, CZ);
  };
  run_etr(z_sp,  e_sp,  Esp,  P_ETR,  wet_sp,  out_z);
  run_etr(z_seq, e_seq, Eseq, P_SETR, wet_seq, out_zs);

  // --- KL outputs (zeros) ---
  fill(out_kl, 0.f, 2);
  (void)n_in; (void)out_size; (void)ws_size;
}
